// LinearGlobalAttention_63917703299768
// MI455X (gfx1250) — matrix operand order repaired and checked by reading, not compiled, not run on hardware
//
#include <hip/hip_runtime.h>
#include <hip/hip_bf16.h>

// ---------------------------------------------------------------------------
// LinearGlobalAttention for MI455X (gfx1250), wave32 + WMMA bf16.
//
// B=8, C=256, N=H*W=16384.  Pipeline (BN folded, proj fused into context):
//   k1: qkv = Wqkv(bf16) @ x          -> q(bf16), kraw(f32), v(bf16)
//   k2: softmax rows of kraw          -> p(bf16, in-place over kraw buffer)
//   k3: ctx[b] = p @ v^T  (atomic f32 K-split)
//   k4: M2[b] = diag(inv) * Wproj * ctx^T  (bf16)
//   k5: out = relu(M2 @ q + bias2)    -> f32 output
// ---------------------------------------------------------------------------

typedef __bf16 bf16;
typedef __attribute__((ext_vector_type(16))) __bf16 v16bf;
typedef __attribute__((ext_vector_type(8)))  float  v8f;

union Frag16 {
    v16bf v;
    struct { uint4 lo, hi; } q;
    bf16 h[16];
    unsigned short s[16];
};

#define WMMA_BF16(A, Bm, Cm) \
    __builtin_amdgcn_wmma_f32_16x16x32_bf16(false, (A), false, (Bm), (short)0, (Cm), false, false)

#define V8F_ZERO ((v8f){0.f,0.f,0.f,0.f,0.f,0.f,0.f,0.f})

// A fragment (16x32 bf16): lane&15 = M row, lane>>4 selects K-halves.
// element e -> K = 16*g + e   (one contiguous 32B run)
__device__ inline v16bf load_frag_a(const bf16* __restrict__ A, int lda,
                                    int m0, int k0, int lane) {
    const int g = lane >> 4;
    const bf16* row = A + (size_t)(m0 + (lane & 15)) * lda + k0;
    Frag16 f;
    f.q.lo = *(const uint4*)(row + 16 * g);
    f.q.hi = *(const uint4*)(row + 16 * g + 8);
    return f.v;
}

// B fragment (32x16 bf16) from a [N][K]-major source (row per output column,
// K contiguous): lane&15 = N column, element e -> K = k0 + 16*g + e.
__device__ inline v16bf load_frag_b_rows(const bf16* __restrict__ B, int ldb,
                                         int n0, int k0, int lane) {
    const int g = lane >> 4;
    const bf16* row = B + (size_t)(n0 + (lane & 15)) * ldb + k0 + 16 * g;
    Frag16 f;
    f.q.lo = *(const uint4*)(row);
    f.q.hi = *(const uint4*)(row + 8);
    return f.v;
}

// B fragment from LDS tile in [n][k] layout, pitch 48 bf16 (96B rows: b128-aligned).
__device__ inline v16bf load_frag_b_lds(const bf16* sB, int ns, int lane) {
    const int g  = lane >> 4;
    const int nn = ns * 16 + (lane & 15);
    Frag16 f;
    f.q.lo = *(const uint4*)&sB[nn * 48 + 16 * g];
    f.q.hi = *(const uint4*)&sB[nn * 48 + 16 * g + 8];
    return f.v;
}

__device__ inline void load8f(const float* __restrict__ p, float* d) {
    const float4 t0 = ((const float4*)p)[0];
    const float4 t1 = ((const float4*)p)[1];
    d[0]=t0.x; d[1]=t0.y; d[2]=t0.z; d[3]=t0.w;
    d[4]=t1.x; d[5]=t1.y; d[6]=t1.z; d[7]=t1.w;
}

// ---------------------------------------------------------------------------
// k_prep: Wqkv f32->bf16; fold BN into inv[] / bias2[].
__global__ __launch_bounds__(256) void k_prep(const float* __restrict__ wq,
                                              bf16* __restrict__ wqb,
                                              const float* __restrict__ gamma,
                                              const float* __restrict__ beta,
                                              const float* __restrict__ rmean,
                                              const float* __restrict__ rvar,
                                              float* __restrict__ inv,
                                              float* __restrict__ bias2) {
    const int i = blockIdx.x * 256 + threadIdx.x;
    if (i < 768 * 256) wqb[i] = (bf16)wq[i];
    if (i < 256) {
        const float iv = gamma[i] * rsqrtf(rvar[i] + 1e-5f);
        inv[i]   = iv;
        bias2[i] = beta[i] - rmean[i] * iv;
    }
}

// ---------------------------------------------------------------------------
// k_qkv: per batch, [768,256]bf16 @ [256,16384] -> q/k/v.
// Block: 256 threads (8 waves). Tile: M=128 (16 per wave), N=128, K-step 32.
// Double-buffered LDS B tile; packed pair ds_store_b32; pipelined B frags.
__global__ __launch_bounds__(256) void k_qkv(const float* __restrict__ x,
                                             const bf16* __restrict__ wqb,
                                             bf16* __restrict__ qb,
                                             float* __restrict__ kraw,
                                             bf16* __restrict__ vb) {
    __shared__ bf16 sB[2][128 * 48];           // [n][k] layout, pitch 48
    const int n0   = blockIdx.x * 128;
    const int m0   = blockIdx.y * 128;
    const int b    = blockIdx.z;
    const int tid  = threadIdx.x;
    const int lane = tid & 31;
    const int w    = tid >> 5;
    const int g    = lane >> 4;

    // cooperative-load mapping: thread owns k-pair kp (k = 2kp, 2kp+1), 8 n's
    const int kp = tid & 15;
    const int nb = (tid >> 4) * 8;
    const float* xt = x + (size_t)b * 256 * 16384 + (size_t)(2 * kp) * 16384 + n0 + nb;

    v8f acc[8];
#pragma unroll
    for (int i = 0; i < 8; ++i) acc[i] = V8F_ZERO;

    float lo[8], hi[8];
    load8f(xt, lo);                    // tile 0, row 2kp
    load8f(xt + 16384, hi);            // tile 0, row 2kp+1

    for (int step = 0; step < 8; ++step) {
        const int cur = step & 1;
        __syncthreads();
        // store tile(step) -> sB[cur] as packed (k,k+1) bf16 pairs
        {
            unsigned int* sb32 = (unsigned int*)&sB[cur][0];
#pragma unroll
            for (int i = 0; i < 8; ++i) {
                union { bf16 h[2]; unsigned int u; } pp;
                pp.h[0] = (bf16)lo[i];
                pp.h[1] = (bf16)hi[i];
                sb32[(nb + i) * 24 + kp] = pp.u;
            }
        }
        float nlo[8], nhi[8];
        if (step < 7) {                // stage tile(step+1) while computing
            const float* nxt = xt + (size_t)(step + 1) * 32 * 16384;
            load8f(nxt, nlo);
            load8f(nxt + 16384, nhi);
        }
        __syncthreads();

        const v16bf a = load_frag_a(wqb, 256, m0 + w * 16, step * 32, lane);
        v16bf fc = load_frag_b_lds(sB[cur], 0, lane);
#pragma unroll
        for (int ns = 0; ns < 8; ++ns) {
            v16bf fn;
            if (ns < 7) fn = load_frag_b_lds(sB[cur], ns + 1, lane);
            acc[ns] = WMMA_BF16(a, fc, acc[ns]);
            fc = fn;
        }
#pragma unroll
        for (int i = 0; i < 8; ++i) { lo[i] = nlo[i]; hi[i] = nhi[i]; }
    }

    // epilogue: class (q/k/v) is uniform per block since M-tiles are 128-wide
    const int cls = m0 >> 8;                       // 0=q, 1=k, 2=v
    const int mb  = (m0 & 255) + w * 16 + 8 * g;   // local row base within class
    if (cls == 0) {
#pragma unroll
        for (int ns = 0; ns < 8; ++ns) {
            const int gn = n0 + ns * 16 + (lane & 15);
#pragma unroll
            for (int r = 0; r < 8; ++r)
                qb[((size_t)b * 256 + mb + r) * 16384 + gn] = (bf16)acc[ns][r];
        }
    } else if (cls == 1) {
#pragma unroll
        for (int ns = 0; ns < 8; ++ns) {
            const int gn = n0 + ns * 16 + (lane & 15);
#pragma unroll
            for (int r = 0; r < 8; ++r)
                kraw[((size_t)b * 256 + mb + r) * 16384 + gn] = acc[ns][r];
        }
    } else {
#pragma unroll
        for (int ns = 0; ns < 8; ++ns) {
            const int gn = n0 + ns * 16 + (lane & 15);
#pragma unroll
            for (int r = 0; r < 8; ++r)
                vb[((size_t)b * 256 + mb + r) * 16384 + gn] = (bf16)acc[ns][r];
        }
    }
}

// ---------------------------------------------------------------------------
// k_softmax: one block per (b,c) row of 16384; row in registers; write bf16
// probs in-place into the kraw buffer (bf16 row pitch 32768 elems).
__global__ __launch_bounds__(512) void k_softmax(float* __restrict__ kbuf) {
    __shared__ float red[512];
    const int row = blockIdx.x;
    const int t   = threadIdx.x;
    const float4* src = (const float4*)(kbuf + (size_t)row * 16384);

    float4 rv[8];
    float lmax = -3.0e38f;
#pragma unroll
    for (int j = 0; j < 8; ++j) {
        rv[j] = src[t + j * 512];
        lmax = fmaxf(lmax, fmaxf(fmaxf(rv[j].x, rv[j].y), fmaxf(rv[j].z, rv[j].w)));
    }
    red[t] = lmax; __syncthreads();
    for (int s = 256; s > 0; s >>= 1) {
        if (t < s) red[t] = fmaxf(red[t], red[t + s]);
        __syncthreads();
    }
    const float m = red[0]; __syncthreads();

    float lsum = 0.f;
#pragma unroll
    for (int j = 0; j < 8; ++j) {
        rv[j].x = __expf(rv[j].x - m); rv[j].y = __expf(rv[j].y - m);
        rv[j].z = __expf(rv[j].z - m); rv[j].w = __expf(rv[j].w - m);
        lsum += rv[j].x + rv[j].y + rv[j].z + rv[j].w;
    }
    red[t] = lsum; __syncthreads();
    for (int s = 256; s > 0; s >>= 1) {
        if (t < s) red[t] += red[t + s];
        __syncthreads();
    }
    const float rinv = 1.0f / red[0];

    uint2* dst = (uint2*)((bf16*)kbuf + (size_t)row * 32768);
#pragma unroll
    for (int j = 0; j < 8; ++j) {
        union { bf16 h[4]; uint2 u; } pk;
        pk.h[0] = (bf16)(rv[j].x * rinv); pk.h[1] = (bf16)(rv[j].y * rinv);
        pk.h[2] = (bf16)(rv[j].z * rinv); pk.h[3] = (bf16)(rv[j].w * rinv);
        dst[t + j * 512] = pk.u;
    }
}

// ---------------------------------------------------------------------------
// k_ctx: ctx[b][c][d] += sum_n p[c,n] v[d,n], K split 8 ways (atomics).
// One 16x16 tile per wave; fragments straight from global (contiguous b128s);
// global_prefetch keeps the two HBM streams ahead of the WMMA loop.
__global__ __launch_bounds__(256) void k_ctx(const bf16* __restrict__ p,   // pitch 32768
                                             const bf16* __restrict__ vb,  // pitch 16384
                                             float* __restrict__ ctx) {
    const int lane  = threadIdx.x & 31;
    const int w     = threadIdx.x >> 5;
    const int t     = blockIdx.x * 8 + w;            // 0..255 tile id
    const int tileM = (t >> 4) * 16;
    const int tileN = (t & 15) * 16;
    const int b     = blockIdx.y;
    const int kbase = blockIdx.z * 2048;

    const bf16* pa   = p  + (size_t)b * 256 * 32768;
    const bf16* vbb  = vb + (size_t)b * 256 * 16384;
    const bf16* arow = pa  + (size_t)(tileM + (lane & 15)) * 32768;
    const bf16* brow = vbb + (size_t)(tileN + (lane & 15)) * 16384;

    v8f acc = V8F_ZERO;
#pragma unroll 4
    for (int k0 = kbase; k0 < kbase + 2048; k0 += 32) {
        __builtin_prefetch(arow + k0 + 512, 0, 0);   // global_prefetch_b8
        __builtin_prefetch(brow + k0 + 512, 0, 0);
        const v16bf a  = load_frag_a(pa, 32768, tileM, k0, lane);
        const v16bf bm = load_frag_b_rows(vbb, 16384, tileN, k0, lane);
        acc = WMMA_BF16(a, bm, acc);
    }
    const int g = lane >> 4;
    float* dst = ctx + (size_t)b * 65536 + (size_t)(tileM + 8 * g) * 256
                     + tileN + (lane & 15);
#pragma unroll
    for (int r = 0; r < 8; ++r) atomicAdd(dst + (size_t)r * 256, acc[r]);
}

// ---------------------------------------------------------------------------
// k_m2: M2[b][o][c] = inv[o] * sum_d Wproj[o][d] * ctx[b][c][d]  (bf16 out)
__global__ __launch_bounds__(256) void k_m2(const float* __restrict__ wproj,
                                            const float* __restrict__ ctx,
                                            const float* __restrict__ inv,
                                            bf16* __restrict__ m2) {
    __shared__ float wrow[256];
    const int b = blockIdx.x >> 8;
    const int o = blockIdx.x & 255;
    const int c = threadIdx.x;
    wrow[c] = wproj[o * 256 + c];
    __syncthreads();
    const float* crow = ctx + (size_t)b * 65536 + (size_t)c * 256;
    float s = 0.f;
#pragma unroll 8
    for (int d = 0; d < 256; ++d) s += wrow[d] * crow[d];
    m2[(size_t)b * 65536 + (size_t)o * 256 + c] = (bf16)(inv[o] * s);
}

// ---------------------------------------------------------------------------
// k_out: out[b] = relu(M2[b] @ q[b] + bias2).  M=256,K=256,N=16384.
// Same double-buffered structure as k_qkv; q is already bf16 so the tile
// transpose packs two source rows into dword LDS stores.
__global__ __launch_bounds__(256) void k_out(const bf16* __restrict__ m2,
                                             const bf16* __restrict__ qb,
                                             const float* __restrict__ bias2,
                                             float* __restrict__ out) {
    __shared__ bf16 sB[2][128 * 48];
    const int n0   = blockIdx.x * 128;
    const int m0   = blockIdx.y * 128;
    const int b    = blockIdx.z;
    const int tid  = threadIdx.x;
    const int lane = tid & 31;
    const int w    = tid >> 5;
    const int g    = lane >> 4;

    const bf16* A  = m2 + (size_t)b * 65536;
    const int  kp  = tid & 15;
    const int  nb  = (tid >> 4) * 8;
    const bf16* qt = qb + ((size_t)b * 256 + 2 * kp) * 16384 + n0 + nb;

    v8f acc[8];
#pragma unroll
    for (int i = 0; i < 8; ++i) acc[i] = V8F_ZERO;

    Frag16 r0, r1;                     // 8 bf16 from row 2kp / 2kp+1
    r0.q.lo = *(const uint4*)(qt);
    r1.q.lo = *(const uint4*)(qt + 16384);

    for (int step = 0; step < 8; ++step) {
        const int cur = step & 1;
        __syncthreads();
        {
            unsigned int* sb32 = (unsigned int*)&sB[cur][0];
#pragma unroll
            for (int i = 0; i < 8; ++i)
                sb32[(nb + i) * 24 + kp] =
                    (unsigned int)r0.s[i] | ((unsigned int)r1.s[i] << 16);
        }
        Frag16 n0f, n1f;
        if (step < 7) {
            const bf16* nxt = qt + (size_t)(step + 1) * 32 * 16384;
            n0f.q.lo = *(const uint4*)(nxt);
            n1f.q.lo = *(const uint4*)(nxt + 16384);
        }
        __syncthreads();

        const v16bf a = load_frag_a(A, 256, m0 + w * 16, step * 32, lane);
        v16bf fc = load_frag_b_lds(sB[cur], 0, lane);
#pragma unroll
        for (int ns = 0; ns < 8; ++ns) {
            v16bf fn;
            if (ns < 7) fn = load_frag_b_lds(sB[cur], ns + 1, lane);
            acc[ns] = WMMA_BF16(a, fc, acc[ns]);
            fc = fn;
        }
        r0 = n0f; r1 = n1f;
    }

    const int gmBase = m0 + w * 16 + 8 * g;
#pragma unroll
    for (int ns = 0; ns < 8; ++ns) {
        const int gn = n0 + ns * 16 + (lane & 15);
#pragma unroll
        for (int r = 0; r < 8; ++r) {
            const int gm = gmBase + r;
            out[((size_t)b * 256 + gm) * 16384 + gn] =
                fmaxf(acc[ns][r] + bias2[gm], 0.f);
        }
    }
}

// ---------------------------------------------------------------------------
extern "C" void kernel_launch(void* const* d_in, const int* in_sizes, int n_in,
                              void* d_out, int out_size, void* d_ws, size_t ws_size,
                              hipStream_t stream) {
    const float* x     = (const float*)d_in[0];
    const float* wqkv  = (const float*)d_in[1];
    const float* wproj = (const float*)d_in[2];
    const float* gamma = (const float*)d_in[3];
    const float* beta  = (const float*)d_in[4];
    const float* rmean = (const float*)d_in[5];
    const float* rvar  = (const float*)d_in[6];

    char* ws = (char*)d_ws;
    bf16*  qb    = (bf16*)(ws + 0);                          //  64 MiB
    bf16*  vb    = (bf16*)(ws + ((size_t)64 << 20));         //  64 MiB
    float* kraw  = (float*)(ws + ((size_t)128 << 20));       // 128 MiB (p reuses it)
    float* ctx   = (float*)(ws + ((size_t)256 << 20));       //   2 MiB
    bf16*  m2    = (bf16*)(ws + ((size_t)258 << 20));        //   1 MiB
    bf16*  wqb   = (bf16*)(ws + ((size_t)259 << 20));        // 384 KiB
    float* inv   = (float*)(ws + ((size_t)259 << 20) + (512 << 10));
    float* bias2 = inv + 256;

    k_prep<<<768, 256, 0, stream>>>(wqkv, wqb, gamma, beta, rmean, rvar, inv, bias2);
    k_qkv<<<dim3(128, 6, 8), 256, 0, stream>>>(x, wqb, qb, kraw, vb);
    k_softmax<<<2048, 512, 0, stream>>>(kraw);
    hipMemsetAsync(ctx, 0, (size_t)8 * 256 * 256 * 4, stream);
    k_ctx<<<dim3(32, 8, 8), 256, 0, stream>>>((const bf16*)kraw, vb, ctx);
    k_m2<<<2048, 256, 0, stream>>>(wproj, ctx, inv, m2);
    k_out<<<dim3(128, 2, 8), 256, 0, stream>>>(m2, qb, bias2, (float*)d_out);
}
